// FCGCN_31997506355256
// MI455X (gfx1250) — matrix operand order repaired and checked by reading, not compiled, not run on hardware
//
#include <hip/hip_runtime.h>
#include <hip/hip_bf16.h>

// ---------------------------------------------------------------------------
// FC-GCN pipeline for gfx1250 (MI455X), bf16 WMMA for all matmuls.
// GEMM: 128x128 block tile, BK=32, 8 waves (wave32) each computing 32x64,
// LDS double-buffered, async global->LDS staging for the A tile
// (GLOBAL_LOAD_ASYNC_TO_LDS_B128, ASYNCcnt), b128 sync staging + 16-bit
// transpose scatter for the B tile, v_wmma_f32_16x16x32_bf16 compute.
// Precondition (all call sites): M % 128 == 0, K % 32 == 0.
// ---------------------------------------------------------------------------

#define USE_ASYNC_STAGE 1

typedef unsigned short u16;
typedef __attribute__((ext_vector_type(16))) __bf16 v16bf;
typedef __attribute__((ext_vector_type(8)))  float  v8f;

#define BM 128
#define BN 128
#define BK 32
#define LDSA 40   // row stride (bf16 elems): 80B -> 16B-aligned rows, conflict-free
#define LDSB 40

#define CN 65536
#define CD_IN 256
#define CH1 512
#define CH2 1024
#define CH 1024
#define CD_OUT 16
#define EPS_BN 1e-5f
#define EPS_COS 1e-10f

enum { EPI_F32 = 0, EPI_BIAS_F32 = 1, EPI_BIAS_RELU_BF16 = 2,
       EPI_BF16 = 3, EPI_RELU_BF16 = 4, EPI_RELU_F32 = 5 };

__device__ __forceinline__ u16 f2bf(float f) {
    unsigned u = __float_as_uint(f);
    unsigned r = u + 0x7fffu + ((u >> 16) & 1u);   // round to nearest even
    return (u16)(r >> 16);
}

union U16x8 { v16bf v; unsigned u[8]; };
union F8    { v8f   v; float    f[8]; };

// ---------------------------------------------------------------------------
// Tiled bf16 WMMA GEMM:  C[M,Ncols] = A[M,K] * B[K,Ncols] (+bias) (+relu)
// TRANSA: A points to H stored [K, M] row-major; computes H^T * B.
// ---------------------------------------------------------------------------
template <int EPI, bool TRANSA>
__global__ __launch_bounds__(256) void gemm_wmma_bf16(
    const u16* __restrict__ A, const u16* __restrict__ B,
    const float* __restrict__ bias, void* __restrict__ Cout,
    int M, int Ncols, int K, int lda, int ldb, int ldc) {

    __shared__ u16 As[2][BM * LDSA];   // [m][k]
    __shared__ u16 Bs[2][BN * LDSB];   // [n][k] (transposed on stage-in)

    const int tid  = threadIdx.x;
    const int m0   = blockIdx.x * BM;
    const int n0   = blockIdx.y * BN;
    const int wave = tid >> 5;
    const int lane = tid & 31;
    const int wm   = (wave & 3) << 5;     // 0,32,64,96
    const int wn   = (wave >> 2) << 6;    // 0,64
    const int l15  = lane & 15;
    const int half = lane >> 4;

    // ---- tile staging ----
    auto stageA = [&](int k0, u16* dst) {
        if (!TRANSA) {
            // 128 rows x 32 bf16 = 512 b128 transfers; 2 per thread
            #pragma unroll
            for (int i = 0; i < 2; ++i) {
                int lin  = tid + i * 256;        // 0..511
                int r    = lin >> 2;             // tile row 0..127
                int quad = lin & 3;              // 8 bf16 each
                const u16* src = A + (size_t)(m0 + r) * lda + k0 + quad * 8;
#if USE_ASYNC_STAGE
                unsigned ldsaddr =
                    (unsigned)(size_t)(dst + r * LDSA + quad * 8);
                unsigned long long gaddr = (unsigned long long)(size_t)src;
                asm volatile("global_load_async_to_lds_b128 %0, %1, off"
                             :: "v"(ldsaddr), "v"(gaddr) : "memory");
#else
                *(uint4*)(dst + r * LDSA + quad * 8) = *(const uint4*)src;
#endif
            }
        } else {
            // tile = H[k0..k0+32, m0..m0+128]: 32 rows x 128 bf16 = 512 b128; 2/thread
            #pragma unroll
            for (int i = 0; i < 2; ++i) {
                int lin  = tid + i * 256;        // 0..511
                int r    = lin >> 4;             // k row 0..31
                int quad = lin & 15;             // m col quad*8
                uint4 v = *(const uint4*)(A + (size_t)(k0 + r) * lda + m0 + quad * 8);
                unsigned w[4] = {v.x, v.y, v.z, v.w};
                #pragma unroll
                for (int e = 0; e < 4; ++e) {
                    int mc = quad * 8 + e * 2;
                    dst[(mc + 0) * LDSA + r] = (u16)(w[e] & 0xffffu);
                    dst[(mc + 1) * LDSA + r] = (u16)(w[e] >> 16);
                }
            }
        }
    };
    auto stageB = [&](int k0, u16* dst) {
        // 32 rows x 128 bf16 = 512 b128 loads; 2 per thread; scatter-transpose
        #pragma unroll
        for (int i = 0; i < 2; ++i) {
            int lin  = tid + i * 256;            // 0..511
            int r    = lin >> 4;                 // k row 0..31
            int quad = lin & 15;                 // n col quad*8
            int gn   = n0 + quad * 8;
            uint4 v = {0u, 0u, 0u, 0u};
            if (gn + 8 <= Ncols)                 // Ncols always multiple of 8
                v = *(const uint4*)(B + (size_t)(k0 + r) * ldb + gn);
            unsigned w[4] = {v.x, v.y, v.z, v.w};
            #pragma unroll
            for (int e = 0; e < 4; ++e) {
                int nc = quad * 8 + e * 2;
                dst[(nc + 0) * LDSB + r] = (u16)(w[e] & 0xffffu);
                dst[(nc + 1) * LDSB + r] = (u16)(w[e] >> 16);
            }
        }
    };
    auto waitStage = [&]() {
#if USE_ASYNC_STAGE
        if (!TRANSA)
            asm volatile("s_wait_asynccnt 0x0" ::: "memory");
#endif
        __syncthreads();
    };

    v8f acc[2][4] = {};
    const int NK = K / BK;

    stageA(0, As[0]);
    stageB(0, Bs[0]);
    waitStage();

    for (int kt = 0; kt < NK; ++kt) {
        const int cur = kt & 1;
        if (kt + 1 < NK) {                       // prefetch next tile into other buf
            stageA((kt + 1) * BK, As[cur ^ 1]);
            stageB((kt + 1) * BK, Bs[cur ^ 1]);
        }

        // ---- build fragments (contiguous k = 16*half + s in both A and B) and issue WMMAs ----
        v16bf afr[2], bfr[4];
        #pragma unroll
        for (int mi = 0; mi < 2; ++mi) {
            U16x8 t;
            const u16* ap = &As[cur][(wm + mi * 16 + l15) * LDSA];
            #pragma unroll
            for (int p = 0; p < 8; ++p) {
                int kb = 2 * p + half * 16;
                t.u[p] = *(const unsigned*)(ap + kb);
            }
            afr[mi] = t.v;
        }
        #pragma unroll
        for (int ni = 0; ni < 4; ++ni) {
            U16x8 t;
            const u16* bp = &Bs[cur][(wn + ni * 16 + l15) * LDSB + half * 16];
            #pragma unroll
            for (int p = 0; p < 8; ++p)
                t.u[p] = *(const unsigned*)(bp + 2 * p);
            bfr[ni] = t.v;
        }
        #pragma unroll
        for (int mi = 0; mi < 2; ++mi)
            #pragma unroll
            for (int ni = 0; ni < 4; ++ni)
                acc[mi][ni] = __builtin_amdgcn_wmma_f32_16x16x32_bf16(
                    false, afr[mi], false, bfr[ni], (short)0, acc[mi][ni],
                    false, false);
        waitStage();
    }

    // ---- epilogue (M % 128 == 0: no row guard needed) ----
    #pragma unroll
    for (int mi = 0; mi < 2; ++mi) {
        #pragma unroll
        for (int ni = 0; ni < 4; ++ni) {
            F8 r; r.v = acc[mi][ni];
            int gn = n0 + wn + ni * 16 + l15;
            bool nok = gn < Ncols;
            #pragma unroll
            for (int rr = 0; rr < 8; ++rr) {
                int gm = m0 + wm + mi * 16 + rr + half * 8;
                float v = r.f[rr];
                if (EPI == EPI_BIAS_F32 || EPI == EPI_BIAS_RELU_BF16)
                    v += nok ? bias[gn] : 0.0f;
                if (EPI == EPI_BIAS_RELU_BF16 || EPI == EPI_RELU_BF16 ||
                    EPI == EPI_RELU_F32) v = fmaxf(v, 0.0f);
                if (nok) {
                    if (EPI == EPI_BIAS_RELU_BF16 || EPI == EPI_BF16 ||
                        EPI == EPI_RELU_BF16)
                        ((u16*)Cout)[(size_t)gm * ldc + gn] = f2bf(v);
                    else
                        ((float*)Cout)[(size_t)gm * ldc + gn] = v;
                }
            }
        }
    }
}

// ---------------------------------------------------------------------------
// Helper kernels
// ---------------------------------------------------------------------------
__global__ void f2bf_kernel(const float* __restrict__ in, u16* __restrict__ out,
                            long long n) {
    long long i = (long long)blockIdx.x * blockDim.x + threadIdx.x;
    long long stride = (long long)gridDim.x * blockDim.x;
    for (; i < n; i += stride) out[i] = f2bf(in[i]);
}

// column mean/var over [Nrows, C]: block = 64 cols x 4 row-groups
__global__ __launch_bounds__(256) void bn_stats_kernel(
    const float* __restrict__ Y, float* __restrict__ mu,
    float* __restrict__ var, int Nrows, int C) {
    __shared__ float ssum[256], ssq[256];
    int t   = threadIdx.x;
    int col = blockIdx.x * 64 + (t & 63);
    int rg  = t >> 6;
    float s = 0.0f, q = 0.0f;
    for (int r = rg; r < Nrows; r += 4) {
        float v = Y[(size_t)r * C + col];
        s += v; q += v * v;
    }
    ssum[t] = s; ssq[t] = q;
    __syncthreads();
    if (rg == 0) {
        s = ssum[t] + ssum[t + 64] + ssum[t + 128] + ssum[t + 192];
        q = ssq[t] + ssq[t + 64] + ssq[t + 128] + ssq[t + 192];
        float m = s / (float)Nrows;
        mu[col]  = m;
        var[col] = q / (float)Nrows - m * m;
    }
}

__global__ void bn_apply_relu_kernel(
    const float* __restrict__ Y, const float* __restrict__ mu,
    const float* __restrict__ var, const float* __restrict__ g,
    const float* __restrict__ be, u16* __restrict__ out,
    long long total, int C) {
    long long i = (long long)blockIdx.x * blockDim.x + threadIdx.x;
    long long stride = (long long)gridDim.x * blockDim.x;
    for (; i < total; i += stride) {
        int c = (int)(i % C);
        float v = (Y[i] - mu[c]) * rsqrtf(var[c] + EPS_BN) * g[c] + be[c];
        out[i] = f2bf(fmaxf(v, 0.0f));
    }
}

__global__ void diag_norm_kernel(const float* __restrict__ G,
                                 float* __restrict__ norms, int H) {
    int i = blockIdx.x * blockDim.x + threadIdx.x;
    if (i < H) norms[i] = sqrtf(fmaxf(G[(size_t)i * H + i], 0.0f));
}

__global__ __launch_bounds__(256) void row_degree_kernel(
    const float* __restrict__ G, const float* __restrict__ norms,
    float* __restrict__ dinv, int H) {
    __shared__ float s[256];
    int i = blockIdx.x;
    float ni = norms[i];
    float acc = 0.0f;
    for (int j = threadIdx.x; j < H; j += 256)
        acc += G[(size_t)i * H + j] / (ni * norms[j] + EPS_COS);
    s[threadIdx.x] = acc;
    __syncthreads();
    for (int st = 128; st > 0; st >>= 1) {
        if (threadIdx.x < st) s[threadIdx.x] += s[threadIdx.x + st];
        __syncthreads();
    }
    if (threadIdx.x == 0) {
        float d = s[0] + 1.0f;                     // +I contribution
        dinv[i] = (d > 0.0f) ? rsqrtf(d) : 0.0f;
    }
}

__global__ void make_S_kernel(const float* __restrict__ G,
                              const float* __restrict__ norms,
                              const float* __restrict__ dinv,
                              u16* __restrict__ S, int H) {
    long long idx = (long long)blockIdx.x * blockDim.x + threadIdx.x;
    long long tot = (long long)H * H;
    if (idx >= tot) return;
    int i = (int)(idx / H), j = (int)(idx % H);
    float a = G[idx] / (norms[i] * norms[j] + EPS_COS) + (i == j ? 1.0f : 0.0f);
    S[idx] = f2bf(dinv[i] * a * dinv[j]);
}

// ---------------------------------------------------------------------------
// Launcher
// ---------------------------------------------------------------------------
extern "C" void kernel_launch(void* const* d_in, const int* in_sizes, int n_in,
                              void* d_out, int out_size, void* d_ws, size_t ws_size,
                              hipStream_t stream) {
    const float* x   = (const float*)d_in[0];
    const float* W1  = (const float*)d_in[1];
    const float* b1  = (const float*)d_in[2];
    const float* g1  = (const float*)d_in[3];
    const float* be1 = (const float*)d_in[4];
    const float* W2  = (const float*)d_in[5];
    const float* b2  = (const float*)d_in[6];
    const float* g2  = (const float*)d_in[7];
    const float* be2 = (const float*)d_in[8];
    const float* W3  = (const float*)d_in[9];
    const float* b3  = (const float*)d_in[10];
    const float* GW1 = (const float*)d_in[11];
    const float* GW2 = (const float*)d_in[12];

    char* ws = (char*)d_ws;
    size_t off = 0;
    auto alloc = [&](size_t bytes) { char* p = ws + off; off += (bytes + 255) & ~(size_t)255; return p; };

    // Region A: 256MB (Y f32, later reused for H3b / O1b / T2b)
    char* regA = alloc((size_t)CN * CH2 * 4);
    // Region B: 32MB (Xb, later reused for G + S)
    char* regB = alloc((size_t)CN * CD_IN * 2);
    // Region C: 64MB (H1b)
    char* regC = alloc((size_t)CN * CH1 * 2);
    // Region D: 128MB (H2b, later reused for T1b)
    char* regD = alloc((size_t)CN * CH2 * 2);
    // Weights (bf16)
    u16* W1b  = (u16*)alloc((size_t)CD_IN * CH1 * 2);
    u16* W2b  = (u16*)alloc((size_t)CH1 * CH2 * 2);
    u16* W3b  = (u16*)alloc((size_t)CH2 * CH * 2);
    u16* GW1b = (u16*)alloc((size_t)CH * CH * 2);
    u16* GW2b = (u16*)alloc((size_t)CH * CD_OUT * 2);
    // Small vectors
    float* mu1   = (float*)alloc(CH1 * 4);
    float* var1  = (float*)alloc(CH1 * 4);
    float* mu2   = (float*)alloc(CH2 * 4);
    float* var2  = (float*)alloc(CH2 * 4);
    float* norms = (float*)alloc(CH * 4);
    float* dinv  = (float*)alloc(CH * 4);

    float* Y   = (float*)regA;
    u16*  H3b  = (u16*)regA;                                    // after Y dead
    u16*  O1b  = (u16*)(regA + (size_t)CN * CH * 2);            // second half of A
    u16*  T2b  = (u16*)regA;                                    // after H3b dead
    u16*  Xb   = (u16*)regB;
    float* G   = (float*)regB;                                  // after Xb dead
    u16*  Sb   = (u16*)(regB + ((size_t)8 << 20));              // past G (4MB)
    u16*  H1b  = (u16*)regC;
    u16*  H2b  = (u16*)regD;
    u16*  T1b  = (u16*)regD;                                    // after H2b dead

    dim3 blk(256);
    // 1. convert inputs/weights to bf16
    f2bf_kernel<<<2048, blk, 0, stream>>>(x,   Xb,   (long long)CN * CD_IN);
    f2bf_kernel<<<256,  blk, 0, stream>>>(W1,  W1b,  (long long)CD_IN * CH1);
    f2bf_kernel<<<512,  blk, 0, stream>>>(W2,  W2b,  (long long)CH1 * CH2);
    f2bf_kernel<<<1024, blk, 0, stream>>>(W3,  W3b,  (long long)CH2 * CH);
    f2bf_kernel<<<1024, blk, 0, stream>>>(GW1, GW1b, (long long)CH * CH);
    f2bf_kernel<<<64,   blk, 0, stream>>>(GW2, GW2b, (long long)CH * CD_OUT);

    // 2. Y = x @ W1 + b1                    [N, 512] f32
    gemm_wmma_bf16<EPI_BIAS_F32, false><<<dim3(CN / BM, CH1 / BN), blk, 0, stream>>>(
        Xb, W1b, b1, Y, CN, CH1, CD_IN, CD_IN, CH1, CH1);
    // 3/4. BN1 + relu -> H1b bf16
    bn_stats_kernel<<<CH1 / 64, blk, 0, stream>>>(Y, mu1, var1, CN, CH1);
    bn_apply_relu_kernel<<<65536, blk, 0, stream>>>(Y, mu1, var1, g1, be1, H1b,
                                                    (long long)CN * CH1, CH1);
    // 5. Y = H1 @ W2 + b2                   [N, 1024] f32
    gemm_wmma_bf16<EPI_BIAS_F32, false><<<dim3(CN / BM, CH2 / BN), blk, 0, stream>>>(
        H1b, W2b, b2, Y, CN, CH2, CH1, CH1, CH2, CH2);
    // 6/7. BN2 + relu -> H2b bf16
    bn_stats_kernel<<<CH2 / 64, blk, 0, stream>>>(Y, mu2, var2, CN, CH2);
    bn_apply_relu_kernel<<<65536, blk, 0, stream>>>(Y, mu2, var2, g2, be2, H2b,
                                                    (long long)CN * CH2, CH2);
    // 8. H3 = relu(H2 @ W3 + b3) -> bf16    [N, 1024] (overwrites Y region)
    gemm_wmma_bf16<EPI_BIAS_RELU_BF16, false><<<dim3(CN / BM, CH / BN), blk, 0, stream>>>(
        H2b, W3b, b3, H3b, CN, CH, CH2, CH2, CH, CH);
    // 9. Gram G = H3^T @ H3                 [1024, 1024] f32 (overwrites Xb region)
    gemm_wmma_bf16<EPI_F32, true><<<dim3(CH / BM, CH / BN), blk, 0, stream>>>(
        H3b, H3b, nullptr, G, CH, CH, CN, CH, CH, CH);
    // 10-12. build normalized support S (bf16)
    diag_norm_kernel<<<(CH + 255) / 256, blk, 0, stream>>>(G, norms, CH);
    row_degree_kernel<<<CH, blk, 0, stream>>>(G, norms, dinv, CH);
    make_S_kernel<<<(CH * CH) / 256, blk, 0, stream>>>(G, norms, dinv, Sb, CH);
    // 13. T1 = H3 @ S -> bf16               (into old H2b region)
    gemm_wmma_bf16<EPI_BF16, false><<<dim3(CN / BM, CH / BN), blk, 0, stream>>>(
        H3b, Sb, nullptr, T1b, CN, CH, CH, CH, CH, CH);
    // 14. O1 = relu(T1 @ GW1) -> bf16       (second half of region A)
    gemm_wmma_bf16<EPI_RELU_BF16, false><<<dim3(CN / BM, CH / BN), blk, 0, stream>>>(
        T1b, GW1b, nullptr, O1b, CN, CH, CH, CH, CH, CH);
    // 15. T2 = O1 @ S -> bf16               (first half of region A; H3 dead)
    gemm_wmma_bf16<EPI_BF16, false><<<dim3(CN / BM, CH / BN), blk, 0, stream>>>(
        O1b, Sb, nullptr, T2b, CN, CH, CH, CH, CH, CH);
    // 16. out = relu(T2 @ GW2) -> f32 d_out [N, 16]
    gemm_wmma_bf16<EPI_RELU_F32, false><<<dim3(CN / BM, 1), blk, 0, stream>>>(
        T2b, GW2b, nullptr, d_out, CN, CD_OUT, CH, CH, CD_OUT, CD_OUT);

    (void)in_sizes; (void)n_in; (void)out_size; (void)ws_size;
}
